// RnnSin_3633542333224
// MI455X (gfx1250) — matrix operand order repaired and checked by reading, not compiled, not run on hardware
//
#include <hip/hip_runtime.h>

// CDNA5 / gfx1250 — wave32, WMMA f16 -> f32.
typedef _Float16 v16h __attribute__((ext_vector_type(16)));
typedef float    v8f  __attribute__((ext_vector_type(8)));
typedef _Float16 h2   __attribute__((ext_vector_type(2)));

static __device__ __forceinline__ h2 pkrtz(float a, float b) {
    // v_cvt_pkrtz_f16_f32
    return __builtin_bit_cast(h2, __builtin_amdgcn_cvt_pkrtz(a, b));
}
// Single-instruction ReLU (avoids clang's canonicalize+max pair for fmaxf).
static __device__ __forceinline__ float relu1(float x) {
    float r;
    asm("v_max_num_f32 %0, 0, %1" : "=v"(r) : "v"(x));
    return r;
}
// Packed f16 ReLU: one VOP3P covers two h-values.
static __device__ __forceinline__ h2 pkmax0(h2 x) {
    h2 r;
    asm("v_pk_max_num_f16 %0, %1, 0" : "=v"(r) : "v"(x));
    return r;
}

// B-operand slot helpers. v16h element 2i/2i+1 = VGPR i lo/hi.
// lo trio -> elems 0..5  : K=0..5  (lanes 0-15) / K=8..13 (lanes 16-31)
// hi trio -> elems 8..13 : K=16..21 (lanes 0-15) / K=24..29 (lanes 16-31)
// All other K positions multiply zero A-coefficients; they are zeroed once
// (must not be NaN/Inf: WMMA 0*Inf -> NaN) and never rewritten.
static __device__ __forceinline__ void set_lo(v16h& v, h2 a, h2 b, h2 c) {
    v[0] = a.x; v[1] = a.y; v[2] = b.x; v[3] = b.y; v[4] = c.x; v[5] = c.y;
}
static __device__ __forceinline__ void set_hi(v16h& v, h2 a, h2 b, h2 c) {
    v[8] = a.x; v[9] = a.y; v[10] = b.x; v[11] = b.y; v[12] = c.x; v[13] = c.y;
}

// Process NS independent 32-element chunks in lockstep so the scheduler can
// fill WMMA->VALU hazard windows with the other stream's work.
// vb/vo are DOUBLE-BUFFERED: the buffer consumed by WMMA t is rewritten only
// after WMMA t+1 has issued (glue depends on D_{t+1} via its RAW-nop chain),
// so no VALU write can land < 4 co-exec slots after a WMMA that reads it —
// the inline-asm writes are hazard-safe by construction.
template <int NS>
static __device__ __forceinline__ void run_chunks(
    const float* __restrict__ X, float* __restrict__ out,
    const long* base, int lane, const v16h& aR, const v16h* aO, float bias)
{
    const v8f czero = {};
    float xv[NS][10];
#pragma unroll
    for (int s = 0; s < NS; ++s) {
        const float2* xp = reinterpret_cast<const float2*>(X + (base[s] + lane) * 10);
#pragma unroll
        for (int i = 0; i < 5; ++i) {          // merged into b64/b128 vmem loads
            const float2 v = xp[i];
            xv[s][2 * i] = v.x; xv[s][2 * i + 1] = v.y;
        }
    }

    v8f  hD[NS], acc[NS];
    v16h vb[NS][2], vo[NS][2];                  // persistent, zero-once operands
#pragma unroll
    for (int s = 0; s < NS; ++s) {
        hD[s] = czero; acc[s] = czero;
        vb[s][0] = (v16h){}; vb[s][1] = (v16h){};
        vo[s][0] = (v16h){}; vo[s][1] = (v16h){};
    }

#pragma unroll
    for (int t = 0; t <= 10; ++t) {
#pragma unroll
        for (int s = 0; s < NS; ++s) {
            // trio_t = f16(relu(h_t)) plus x_t in the augmented slot (K=5).
            const h2    q0 = pkmax0(pkrtz(hD[s][0], hD[s][1]));
            const h2    q1 = pkmax0(pkrtz(hD[s][2], hD[s][3]));
            const float m4 = relu1(hD[s][4]);
            const float xs = (t < 10) ? xv[s][t] : 0.0f;
            const h2    q2 = pkrtz(m4, xs);

            if (t < 10) {                        // recurrence WMMA (alt. buffer)
                v16h& b = vb[s][t & 1];
                set_lo(b, q0, q1, q2);
                hD[s] = __builtin_amdgcn_wmma_f32_16x16x32_f16(
                            false, aR, false, b, (short)0, czero, false, false);
            }
            if (t >= 1) {                        // output layer, eager per pair
                const int p = (t - 1) >> 1;
                v16h& o = vo[s][p & 1];          // alt. buffer per pair
                if (((t - 1) & 1) == 0) {
                    set_lo(o, q0, q1, q2);       // h_{2p+1}
                } else {
                    set_hi(o, q0, q1, q2);       // h_{2p+2} -> pair complete
                    acc[s] = __builtin_amdgcn_wmma_f32_16x16x32_f16(
                                 false, aO[p], false, o, (short)0,
                                 acc[s], false, false);
                }
            }
        }
    }
    // D row0 (lanes 0-15) / row8 (lanes 16-31) both live in VGPR0 of acc.
#pragma unroll
    for (int s = 0; s < NS; ++s) out[base[s] + lane] = acc[s][0] + bias;
}

__global__ __launch_bounds__(256) void rnn_wmma_kernel(
    const float* __restrict__ X,    // [B, 10, 1]
    const float* __restrict__ Wih,  // [5, 1]
    const float* __restrict__ Whh,  // [5, 5]
    const float* __restrict__ Wout, // [1, 50]
    const float* __restrict__ bo,   // [1]
    float* __restrict__ out,        // [B, 1]
    int nChunks)                    // B / 32
{
    const int tid      = blockIdx.x * blockDim.x + threadIdx.x;
    const int wave     = tid >> 5;
    const int lane     = tid & 31;
    const int numWaves = (gridDim.x * blockDim.x) >> 5;
    const h2  zz       = {};

    // ---- A_rnn: rows j=0..4 hold [W_hh(j,:) | W_ih(j)] at K=0..5 (tile 1),
    //      rows 8..12 hold the same at K=8..13 (tile 2).
    // A layout (16-bit 16x32): lane l = row l%16; lanes 0-15 V0-3=K0..7, V4-7=K16..23.
    const bool a1 = (lane < 5);
    const bool a2 = (lane >= 24) && (lane < 29);
    const int  j  = a1 ? lane : (a2 ? (lane - 24) : 0);   // clamped -> in-bounds loads
    const float w0 = Whh[j * 5 + 0], w1 = Whh[j * 5 + 1], w2 = Whh[j * 5 + 2];
    const float w3 = Whh[j * 5 + 3], w4 = Whh[j * 5 + 4];
    const float wi = Wih[j];
    const h2 t0 = pkrtz(w0, w1), t1 = pkrtz(w2, w3), t2 = pkrtz(w4, wi);
    v16h aR = {};
    if (a1) set_lo(aR, t0, t1, t2);
    if (a2) { aR[0] = t0.x; aR[1] = t0.y; aR[2] = t1.x; aR[3] = t1.y; aR[4] = t2.x; aR[5] = t2.y; }

    // ---- A_out[p]: W_out slice for hidden-state pair (h_{2p+1}, h_{2p+2}).
    // Row 0 (tile1): K0..4 = W_out[10p..], K16..20 = W_out[10p+5..]
    //   -> lane 0: lo-trio in elems 0..5, hi-trio in elems 8..13.
    // Row 8 (tile2): K8..12 / K24..28
    //   -> lane 24: lo-trio in elems 0..5, hi-trio in elems 8..13.
    v16h aO[5];
#pragma unroll
    for (int p = 0; p < 5; ++p) {
        float w[10];
#pragma unroll
        for (int i = 0; i < 10; ++i) w[i] = Wout[10 * p + i];
        const h2 l0 = pkrtz(w[0], w[1]), l1 = pkrtz(w[2], w[3]), l2 = pkrtz(w[4], 0.0f);
        const h2 u0 = pkrtz(w[5], w[6]), u1 = pkrtz(w[7], w[8]), u2 = pkrtz(w[9], 0.0f);
        const h2 p0 = (lane == 0 || lane == 24) ? l0 : zz;
        const h2 p1 = (lane == 0 || lane == 24) ? l1 : zz;
        const h2 p2 = (lane == 0 || lane == 24) ? l2 : zz;
        const h2 q0 = (lane == 0 || lane == 24) ? u0 : zz;
        const h2 q1 = (lane == 0 || lane == 24) ? u1 : zz;
        const h2 q2 = (lane == 0 || lane == 24) ? u2 : zz;
        v16h a = {};
        set_lo(a, p0, p1, p2);
        set_hi(a, q0, q1, q2);
        aO[p] = a;
    }

    const float bias = bo[0];

    // ---- Grid-stride over 32-element chunks; EXEC all-ones at every WMMA.
    int c = wave;
    for (; c + numWaves < nChunks; c += 2 * numWaves) {     // dual-stream path
        __builtin_prefetch(X + ((long)(c + 2 * numWaves) * 32 + lane) * 10, 0, 1);
        const long b2[2] = { (long)c * 32, (long)(c + numWaves) * 32 };
        run_chunks<2>(X, out, b2, lane, aR, aO, bias);
    }
    for (; c < nChunks; c += numWaves) {                    // tail (<=1 iter)
        const long b1[1] = { (long)c * 32 };
        run_chunks<1>(X, out, b1, lane, aR, aO, bias);
    }
}

extern "C" void kernel_launch(void* const* d_in, const int* in_sizes, int n_in,
                              void* d_out, int out_size, void* d_ws, size_t ws_size,
                              hipStream_t stream) {
    (void)n_in; (void)d_ws; (void)ws_size; (void)out_size;
    const float* X    = (const float*)d_in[0];
    const float* Wih  = (const float*)d_in[1];
    const float* Whh  = (const float*)d_in[2];
    const float* Wout = (const float*)d_in[3];
    const float* bo   = (const float*)d_in[4];
    float* out = (float*)d_out;

    const int B       = in_sizes[0] / 10;   // X is [B, 10, 1]
    const int nChunks = B / 32;             // 32 batch elems per wave-chunk

    // 4096 waves = 512 blocks x 256 threads for B=1M -> 8 chunks/wave (4 dual iters).
    int totalWaves = (nChunks + 7) / 8; if (totalWaves < 1) totalWaves = 1;
    int blocks = (totalWaves + 7) / 8;  if (blocks < 1) blocks = 1;
    if (blocks > 8192) blocks = 8192;

    rnn_wmma_kernel<<<blocks, 256, 0, stream>>>(X, Wih, Whh, Wout, bo, out, nChunks);
}
